// mfpNet_19258633355273
// MI455X (gfx1250) — matrix operand order repaired and checked by reading, not compiled, not run on hardware
//
#include <hip/hip_runtime.h>
#include <hip/hip_bf16.h>

typedef __attribute__((ext_vector_type(16))) __bf16 v16bf;
typedef __attribute__((ext_vector_type(8)))  float  v8f;
typedef __attribute__((ext_vector_type(4)))  int    v4i_t;

#define D_DIM   128
#define S_SLOTS 8
#define KD_DIM  8
#define KH_DIM  32
#define H_DIM   32
#define OUT_DIM 80
#define KPAD    160   // 130 padded up to 5 chunks of K=32

#if defined(__gfx1250__) && \
    __has_builtin(__builtin_amdgcn_global_load_async_to_lds_b128) && \
    __has_builtin(__builtin_amdgcn_s_wait_asynccnt)
#define USE_ASYNC_LDS 1
#else
#define USE_ASYNC_LDS 0
#endif

// ---------------------------------------------------------------------------
// Kernel 1: per-neighbor key MLP (WMMA bf16) + RBF attention -> atten[N, 8]
//   block = 128 threads = 4 waves; each wave owns a 16-row tile (64 rows/blk)
//   x tile staged fp32 in LDS via async global->LDS copies (ASYNCcnt path),
//   converted to bf16 while packing WMMA A-fragments.
// ---------------------------------------------------------------------------
__global__ __launch_bounds__(128)
void k1_atten(const float* __restrict__ nbrs_enc,
              const float* __restrict__ ref_pos,
              const int*   __restrict__ nbr_src,
              const int*   __restrict__ nbr_seg,
              const float* __restrict__ Wk1, const float* __restrict__ bk1,
              const float* __restrict__ Wk2, const float* __restrict__ bk2,
              const float* __restrict__ slot_keys,
              float* __restrict__ atten_out, int n_total)
{
    __shared__ float  sXf[64][KPAD];         // x tile fp32 (memory format)
    __shared__ __bf16 sW1[KH_DIM][KPAD];     // Wk1 transposed [col][k], bf16
    __shared__ __bf16 sW2[16][KH_DIM];       // Wk2 transposed, cols padded to 16
    __shared__ __bf16 sH[4][16][KH_DIM];     // per-wave hidden activations
    __shared__ float  sKey[4][16][KD_DIM];   // per-wave key features
    __shared__ float  sSlot[S_SLOTS * KD_DIM];

    const int tid     = threadIdx.x;
    const int rowBase = blockIdx.x * 64;

    // --- x tile, k < 128: async global -> LDS (B128 = 4 floats per issue) ---
#if USE_ASYNC_LDS
    for (int idx = tid; idx < 64 * (D_DIM / 4); idx += 128) {
        int r = idx >> 5;              // 32 b128 transfers per row
        int q = (idx & 31) * 4;
        int i = rowBase + r;
        if (i < n_total) {
            __builtin_amdgcn_global_load_async_to_lds_b128(
                (v4i_t*)(nbrs_enc + (long)i * D_DIM + q),
                (v4i_t*)&sXf[r][q], 0, 0);
        } else {
            sXf[r][q] = 0.f; sXf[r][q + 1] = 0.f;
            sXf[r][q + 2] = 0.f; sXf[r][q + 3] = 0.f;
        }
    }
#else
    for (int idx = tid; idx < 64 * D_DIM; idx += 128) {
        int r = idx / D_DIM, k = idx % D_DIM;
        int i = rowBase + r;
        sXf[r][k] = (i < n_total) ? nbrs_enc[(long)i * D_DIM + k] : 0.f;
    }
#endif

    // --- x tile, k in [128, 160): pos_enc gathers + zero padding ---
    for (int idx = tid; idx < 64 * (KPAD - D_DIM); idx += 128) {
        int r = idx >> 5;
        int k = D_DIM + (idx & 31);
        int i = rowBase + r;
        float v = 0.f;
        if (i < n_total && k < D_DIM + 2) {
            int s0 = nbr_src[i], s1 = nbr_seg[i];
            int c2 = k - D_DIM;
            v = ref_pos[s0 * 2 + c2] - ref_pos[s1 * 2 + c2];
        }
        sXf[r][k] = v;
    }

    // --- stage Wk1 (transposed, K zero-padded 130->160) as bf16 ---
    for (int idx = tid; idx < KH_DIM * KPAD; idx += 128) {
        int c = idx / KPAD, k = idx % KPAD;
        float v = (k < D_DIM + 2) ? Wk1[k * KH_DIM + c] : 0.f;
        sW1[c][k] = (__bf16)v;
    }
    // --- stage Wk2 (transposed, cols 8..15 zero) ---
    for (int idx = tid; idx < 16 * KH_DIM; idx += 128) {
        int c = idx / KH_DIM, k = idx % KH_DIM;
        float v = (c < KD_DIM) ? Wk2[k * KD_DIM + c] : 0.f;
        sW2[c][k] = (__bf16)v;
    }
    if (tid < S_SLOTS * KD_DIM) sSlot[tid] = slot_keys[tid];

#if USE_ASYNC_LDS
    __builtin_amdgcn_s_wait_asynccnt(0);   // this wave's async copies landed
#endif
    __syncthreads();                       // all waves' stores/copies visible

    const int wave = tid >> 5;
    const int lane = tid & 31;
    const int mrow = lane & 15;   // A row / B-D col (low half)
    const int half = lane >> 4;   // lane-half selects K sub-range

    // --- layer 1: y[16,32] = x[16,160] @ Wk1[160,32] via 5x2 WMMA ---
    v8f acc0 = {}; v8f acc1 = {};
    for (int c = 0; c < 5; ++c) {
        v16bf a;
        #pragma unroll
        for (int j = 0; j < 8; ++j)
            a[j]     = (__bf16)sXf[wave * 16 + mrow][c * 32 + half * 8 + j];
        #pragma unroll
        for (int j = 0; j < 8; ++j)
            a[8 + j] = (__bf16)sXf[wave * 16 + mrow][c * 32 + 16 + half * 8 + j];
        v16bf b0, b1;
        #pragma unroll
        for (int j = 0; j < 16; ++j) b0[j] = sW1[mrow][c * 32 + half * 8 + j + (j & 8)];
        #pragma unroll
        for (int j = 0; j < 16; ++j) b1[j] = sW1[mrow + 16][c * 32 + half * 8 + j + (j & 8)];
        acc0 = __builtin_amdgcn_wmma_f32_16x16x32_bf16(false, a, false, b0, (short)0, acc0, false, false);
        acc1 = __builtin_amdgcn_wmma_f32_16x16x32_bf16(false, a, false, b1, (short)0, acc1, false, false);
    }

    // --- bias + ReLU in D layout, spill to LDS as bf16 A-operand ---
    #pragma unroll
    for (int j = 0; j < 8; ++j) {
        int r = j + 8 * half;
        float v0 = acc0[j] + bk1[mrow];
        float v1 = acc1[j] + bk1[mrow + 16];
        sH[wave][r][mrow]      = (__bf16)fmaxf(v0, 0.f);
        sH[wave][r][mrow + 16] = (__bf16)fmaxf(v1, 0.f);
    }

    // --- layer 2: key[16,8(pad16)] = h[16,32] @ Wk2[32,16] : one WMMA ---
    v16bf ah, bw;
    #pragma unroll
    for (int j = 0; j < 8; ++j) ah[j]     = sH[wave][mrow][half * 8 + j];
    #pragma unroll
    for (int j = 0; j < 8; ++j) ah[8 + j] = sH[wave][mrow][16 + half * 8 + j];
    #pragma unroll
    for (int j = 0; j < 16; ++j) bw[j]    = sW2[mrow][half * 8 + j + (j & 8)];
    v8f acc2 = {};
    acc2 = __builtin_amdgcn_wmma_f32_16x16x32_bf16(false, ah, false, bw, (short)0, acc2, false, false);

    if (mrow < KD_DIM) {
        #pragma unroll
        for (int j = 0; j < 8; ++j)
            sKey[wave][j + 8 * half][mrow] = acc2[j] + bk2[mrow];
    }

    // --- RBF attention: atten[r,s] = exp(-||key[r] - slot[s]||) ---
    #pragma unroll
    for (int t = 0; t < 4; ++t) {
        int e = lane + 32 * t;      // 0..127 -> (row, slot)
        int r = e >> 3, s = e & 7;
        float d2 = 0.f;
        #pragma unroll
        for (int kd = 0; kd < KD_DIM; ++kd) {
            float df = sKey[wave][r][kd] - sSlot[s * KD_DIM + kd];
            d2 = fmaf(df, df, d2);
        }
        int i = rowBase + wave * 16 + r;
        if (i < n_total)
            atten_out[(long)i * S_SLOTS + s] = __expf(-sqrtf(d2));
    }
}

// ---------------------------------------------------------------------------
// Kernel 2: per-agent segment aggregation (register-resident agg[1024])
//           + 1024->32->32->80 MLP out of LDS.  block = 256 threads / agent.
// ---------------------------------------------------------------------------
__global__ __launch_bounds__(256)
void k2_agg_mlp(const float* __restrict__ nbrs_enc,
                const int*   __restrict__ nbr_seg,
                const float* __restrict__ atten,
                const float* __restrict__ Wh1, const float* __restrict__ bh1,
                const float* __restrict__ Wh2, const float* __restrict__ bh2,
                const float* __restrict__ Wh3, const float* __restrict__ bh3,
                float* __restrict__ out, int n_total)
{
    __shared__ float sAgg[S_SLOTS * D_DIM];   // 1024
    __shared__ float sPart[H_DIM][8];
    __shared__ float sH1[H_DIM];
    __shared__ float sH2[H_DIM];

    const int a   = blockIdx.x;
    const int tid = threadIdx.x;

    // segment bounds via binary search on sorted nbr_seg
    int lo = 0, hi = n_total;
    while (lo < hi) { int m = (lo + hi) >> 1; if (nbr_seg[m] < a) lo = m + 1; else hi = m; }
    const int seg_start = lo;
    hi = n_total;
    while (lo < hi) { int m = (lo + hi) >> 1; if (nbr_seg[m] < a + 1) lo = m + 1; else hi = m; }
    const int seg_end = lo;

    // thread t owns agg elems e = t + 256k  (same d = t&127 for all 4)
    const int d  = tid & 127;
    const int sb = tid >> 7;
    float acc[4] = {0.f, 0.f, 0.f, 0.f};
    for (int i = seg_start; i < seg_end; ++i) {
        if (i + 1 < seg_end)   // hide next row's HBM latency
            __builtin_prefetch(&nbrs_enc[(long)(i + 1) * D_DIM + d], 0, 0);
        float e = nbrs_enc[(long)i * D_DIM + d];
        #pragma unroll
        for (int k = 0; k < 4; ++k)
            acc[k] = fmaf(atten[(long)i * S_SLOTS + sb + 2 * k], e, acc[k]);
    }
    #pragma unroll
    for (int k = 0; k < 4; ++k)
        sAgg[tid + 256 * k] = acc[k];
    __syncthreads();

    // h1 = relu(agg @ Wh1 + bh1): 8-way K-partials per output j
    {
        int j = tid & 31, p = tid >> 5;
        float s = 0.f;
        for (int e = p * 128; e < (p + 1) * 128; ++e)
            s = fmaf(sAgg[e], Wh1[(long)e * H_DIM + j], s);
        sPart[j][p] = s;
    }
    __syncthreads();
    if (tid < H_DIM) {
        float s = bh1[tid];
        #pragma unroll
        for (int p = 0; p < 8; ++p) s += sPart[tid][p];
        sH1[tid] = fmaxf(s, 0.f);
    }
    __syncthreads();
    if (tid < H_DIM) {
        float s = bh2[tid];
        #pragma unroll
        for (int k = 0; k < H_DIM; ++k) s = fmaf(sH1[k], Wh2[k * H_DIM + tid], s);
        sH2[tid] = fmaxf(s, 0.f);
    }
    __syncthreads();
    if (tid < OUT_DIM) {
        float s = bh3[tid];
        #pragma unroll
        for (int k = 0; k < H_DIM; ++k) s = fmaf(sH2[k], Wh3[k * OUT_DIM + tid], s);
        out[(long)a * OUT_DIM + tid] = s;
    }
}

// ---------------------------------------------------------------------------
extern "C" void kernel_launch(void* const* d_in, const int* in_sizes, int n_in,
                              void* d_out, int out_size, void* d_ws, size_t ws_size,
                              hipStream_t stream)
{
    const float* nbrs_enc  = (const float*)d_in[0];
    const float* ref_pos   = (const float*)d_in[1];
    const int*   nbr_src   = (const int*)  d_in[2];
    const int*   nbr_seg   = (const int*)  d_in[3];
    const float* Wk1       = (const float*)d_in[4];
    const float* bk1       = (const float*)d_in[5];
    const float* Wk2       = (const float*)d_in[6];
    const float* bk2       = (const float*)d_in[7];
    const float* slot_keys = (const float*)d_in[8];
    const float* Wh1       = (const float*)d_in[9];
    const float* bh1       = (const float*)d_in[10];
    const float* Wh2       = (const float*)d_in[11];
    const float* bh2       = (const float*)d_in[12];
    const float* Wh3       = (const float*)d_in[13];
    const float* bh3       = (const float*)d_in[14];

    const int N = in_sizes[2];       // flat neighbor count
    const int A = in_sizes[1] / 2;   // ref_pos is [A,2]

    float* atten = (float*)d_ws;     // N * 8 floats of scratch

    const int blocks1 = (N + 63) / 64;
    k1_atten<<<blocks1, 128, 0, stream>>>(nbrs_enc, ref_pos, nbr_src, nbr_seg,
                                          Wk1, bk1, Wk2, bk2, slot_keys,
                                          atten, N);

    k2_agg_mlp<<<A, 256, 0, stream>>>(nbrs_enc, nbr_seg, atten,
                                      Wh1, bh1, Wh2, bh2, Wh3, bh3,
                                      (float*)d_out, N);
}
